// LSTMModelA_40793599377857
// MI455X (gfx1250) — matrix operand order repaired and checked by reading, compiled, not run on hardware
//
#include <hip/hip_runtime.h>

typedef __attribute__((ext_vector_type(16))) _Float16 v16h;
typedef __attribute__((ext_vector_type(8)))  _Float16 v8h;
typedef __attribute__((ext_vector_type(2)))  _Float16 v2h;
typedef __attribute__((ext_vector_type(8)))  float    v8f;
typedef __attribute__((ext_vector_type(4)))  float    v4f;

#define HID     25
#define TSTEPS  240
#define NTILE   7          // ceil(4*HID/16) -> 112 padded gate rows
#define KPAD    32         // K padded: 25 hidden + x@25 + 1.0@26 + zeros

// ---- fast activations: single v_tanh_f32 / v_rcp_f32, no IEEE divides ----
#if __has_builtin(__builtin_amdgcn_tanhf)
__device__ __forceinline__ float ftanh(float x) {
    return __builtin_amdgcn_tanhf(x);        // v_tanh_f32 (TRANS32)
}
#else
__device__ __forceinline__ float ftanh(float x) {
    float t = __expf(-2.0f * fabsf(x));
    float r = (1.0f - t) * __builtin_amdgcn_rcpf(1.0f + t);
    return copysignf(r, x);
}
#endif
__device__ __forceinline__ float fsigmoid(float x) {
    // sigmoid(x) = 0.5 + 0.5*tanh(x/2)
    return fmaf(0.5f, ftanh(0.5f * x), 0.5f);
}

// One wave (32 threads) handles 16 batch rows for all 240 timesteps.
// gates^T = A(W_hh', 112x32) x B(h^T, 32x16)  via v_wmma_f32_16x16x32_f16.
// Gate rows reordered g' = 4*u + type  (type: 0=i,1=f,2=g,3=o) so each lane's
// 8 accumulator rows = 2 hidden units with all four gates -> in-lane cell math.
// Input projection + bias are fused into the WMMA via extra K columns:
// K=25 carries x_t, K=26 carries constant 1.0 (both patched in registers).
__global__ void __launch_bounds__(32)
lstm_fk_wmma(const float* __restrict__ x,
             const float* __restrict__ W_ih,
             const float* __restrict__ W_hh,
             const float* __restrict__ b_ih,
             const float* __restrict__ b_hh,
             const float* __restrict__ W_fc,
             const float* __restrict__ b_fc,
             float* __restrict__ out)
{
    __shared__ _Float16 whh_s[112 * KPAD];  // reordered/padded A matrix, f16
    __shared__ _Float16 hbuf[16 * KPAD];    // h^T staging: [batch][k]

    const int lane = threadIdx.x;           // wave32
    const int bb   = lane & 15;             // batch row within tile (= N column)
    const int half = lane >> 4;             // K-half selector for A/B fragments
    const int bglob = blockIdx.x * 16 + bb; // global batch row

    // ---- Stage A-matrix into LDS: W_hh'[g'][k], g'=4u+type, plus fused
    //      input-projection column (k=25 -> W_ih) and bias column (k=26). ----
    for (int idx = lane; idx < 112 * KPAD; idx += 32) {
        int gp = idx >> 5;         // g' row
        int k  = idx & (KPAD - 1); // K column
        int u  = gp >> 2;
        int ty = gp & 3;
        float v = 0.0f;
        if (u < HID) {
            int g = ty * HID + u;              // PyTorch row order i,f,g,o
            if (k < HID)           v = W_hh[g * HID + k];
            else if (k == HID)     v = W_ih[g];               // x column
            else if (k == HID + 1) v = b_ih[g] + b_hh[g];     // bias column
        }
        whh_s[idx] = (_Float16)v;
    }
    // h^T buffer: h0 = 0 (K>=25 slots are register-patched, don't-care)
    for (int idx = lane; idx < 16 * KPAD; idx += 32)
        hbuf[idx] = (_Float16)0.0f;
    __syncthreads();

    // ---- Load constant A fragments (16-bit A 16x32 layout):
    //      lane: M = lane%16, K runs {8*half..+7} and {16+8*half..+7} ----
    v16h afrag[NTILE];
    #pragma unroll
    for (int t = 0; t < NTILE; ++t) {
        int gp = 16 * t + bb;
        v8h lo = *(const v8h*)&whh_s[gp * KPAD + 8 * half];
        v8h hi = *(const v8h*)&whh_s[gp * KPAD + 16 + 8 * half];
        afrag[t] = __builtin_shufflevector(lo, hi,
                     0,1,2,3,4,5,6,7,8,9,10,11,12,13,14,15);
    }

    // cell state: 2 hidden units per tile per lane (u = 4t + 2*half + {0,1})
    float cst[2 * NTILE];
    #pragma unroll
    for (int i = 0; i < 2 * NTILE; ++i) cst[i] = 0.0f;

    const float* xrow = x + (long)bglob * TSTEPS;
    const _Float16 onef16 = (_Float16)1.0f;

    for (int t0 = 0; t0 < TSTEPS; t0 += 4) {
        v4f xq = *(const v4f*)(xrow + t0);   // 16B-aligned (960|16, 4*t0|16)
        #pragma unroll
        for (int j = 0; j < 4; ++j) {
            float xv = xq[j];

            // B fragment (16-bit B 32x16): lane -> N=bb, K={8*half..+7} and {16+8*half..+7}
            v8h blo = *(const v8h*)&hbuf[bb * KPAD + 8 * half];
            v8h bhi = *(const v8h*)&hbuf[bb * KPAD + 16 + 8 * half];
            v16h bfrag = __builtin_shufflevector(blo, bhi,
                           0,1,2,3,4,5,6,7,8,9,10,11,12,13,14,15);
            // register-patch fused columns (only half-1 lanes carry K=25,26)
            if (half) {
                bfrag[9]  = (_Float16)xv;   // K=25: x_t
                bfrag[10] = onef16;         // K=26: bias multiplier
            }

            #pragma unroll
            for (int t = 0; t < NTILE; ++t) {
                v8f cz = {0.f,0.f,0.f,0.f,0.f,0.f,0.f,0.f};
                // full pre-activation gates (Wx + Wh*h + b) in one WMMA
                v8f d = __builtin_amdgcn_wmma_f32_16x16x32_f16(
                            false, afrag[t], false, bfrag,
                            (short)0, cz, false, false);
                // rows r: u_local = r>>2, type = r&3  (i,f,g,o)
                float hnew[2];
                #pragma unroll
                for (int uu = 0; uu < 2; ++uu) {
                    float gi = fsigmoid(d[4 * uu + 0]);
                    float gf = fsigmoid(d[4 * uu + 1]);
                    float gg = ftanh   (d[4 * uu + 2]);
                    float go = fsigmoid(d[4 * uu + 3]);
                    float cn = fmaf(gf, cst[2 * t + uu], gi * gg);
                    cst[2 * t + uu] = cn;
                    hnew[uu] = go * ftanh(cn);
                }
                // store h for units (4t + 2*half) and (+1), packed f16x2
                v2h hp;
                hp[0] = (_Float16)hnew[0];
                hp[1] = (_Float16)hnew[1];
                *(v2h*)&hbuf[bb * KPAD + 4 * t + 2 * half] = hp;
            }
            __syncthreads();  // single-wave WG: cheap; orders h stores vs loads
        }
    }

    // ---- FC head: logits[b, c] = h . W_fc[c] + b_fc[c], lanes 0..15 ----
    if (half == 0) {
        float a0 = b_fc[0], a1 = b_fc[1];
        #pragma unroll
        for (int u = 0; u < HID; ++u) {
            float hv = (float)hbuf[bb * KPAD + u];
            a0 = fmaf(hv, W_fc[u], a0);
            a1 = fmaf(hv, W_fc[HID + u], a1);
        }
        out[(long)bglob * 2 + 0] = a0;
        out[(long)bglob * 2 + 1] = a1;
    }
}

extern "C" void kernel_launch(void* const* d_in, const int* in_sizes, int n_in,
                              void* d_out, int out_size, void* d_ws, size_t ws_size,
                              hipStream_t stream) {
    const float* x    = (const float*)d_in[0];
    const float* W_ih = (const float*)d_in[1];
    const float* W_hh = (const float*)d_in[2];
    const float* b_ih = (const float*)d_in[3];
    const float* b_hh = (const float*)d_in[4];
    const float* W_fc = (const float*)d_in[5];
    const float* b_fc = (const float*)d_in[6];
    float* out = (float*)d_out;

    const int B = in_sizes[0] / TSTEPS;      // INPUT_SIZE == 1
    const int blocks = B / 16;               // 16 batch rows per wave
    lstm_fk_wmma<<<blocks, 32, 0, stream>>>(x, W_ih, W_hh, b_ih, b_hh,
                                            W_fc, b_fc, out);
}
